// KVAE_47390669144181
// MI455X (gfx1250) — matrix operand order repaired and checked by reading, not compiled, not run on hardware
//
#include <hip/hip_runtime.h>

// ---------------------------------------------------------------------------
// KVAE forward for MI455X (gfx1250, wave32, WMMA).
// conv2/conv3 (99% of FLOPs) -> v_wmma_f32_16x16x32_f16 implicit GEMM.
// Sequential LSTM / Kalman filter / RTS smoother -> small-block f32 VALU.
// ---------------------------------------------------------------------------

typedef _Float16 v16h __attribute__((ext_vector_type(16)));
typedef _Float16 v8h  __attribute__((ext_vector_type(8)));
typedef float    v8f  __attribute__((ext_vector_type(8)));

#define F_TOT 384      // B*T frames
#define T_SEQ 128
#define B_SZ  3

// d_out flat offsets (return order: z_smooth, P_smooth, a_all, A, C, last_z, last_P)
#define OFF_ZS   0
#define OFF_PS   3810
#define OFF_AALL 41910
#define OFF_A    43830
#define OFF_C    82230
#define OFF_LZ   101430
#define OFF_LP   101460

// ---------------------------------------------------------------------------
// Conv1: 64x64x1 -> 32x32x64, k4 s2 pad1, ReLU, f32 compute, f16 out (NHWC)
// ---------------------------------------------------------------------------
__global__ __launch_bounds__(256) void conv1_kernel(
    const float* __restrict__ img, const float* __restrict__ w1,
    const float* __restrict__ b1, _Float16* __restrict__ out) {
  int idx = blockIdx.x * blockDim.x + threadIdx.x;   // F*32*32*64 exact
  int oc = idx & 63;
  int p  = idx >> 6;
  int ox = p & 31; p >>= 5;
  int oy = p & 31; p >>= 5;
  int f  = p;
  if (f >= F_TOT) return;
  float s = b1[oc];
#pragma unroll
  for (int kh = 0; kh < 4; ++kh) {
    int iy = oy * 2 + kh - 1;
    if (iy < 0 || iy >= 64) continue;
#pragma unroll
    for (int kw = 0; kw < 4; ++kw) {
      int ix = ox * 2 + kw - 1;
      if (ix < 0 || ix >= 64) continue;
      s += img[(f * 64 + iy) * 64 + ix] * w1[(kh * 4 + kw) * 64 + oc];
    }
  }
  out[idx] = (_Float16)(s > 0.f ? s : 0.f);
}

// ---------------------------------------------------------------------------
// Weight prep: HWIO f32 -> [Cout][K] f16, K = (kh*4+kw)*CIN + ic
// ---------------------------------------------------------------------------
template <int CIN, int COUT>
__global__ __launch_bounds__(256) void wprep_kernel(
    const float* __restrict__ w, _Float16* __restrict__ wt) {
  int idx = blockIdx.x * blockDim.x + threadIdx.x;  // 16*CIN*COUT exact
  int oc = idx % COUT;
  int k  = idx / COUT;
  wt[(size_t)oc * (16 * CIN) + k] = (_Float16)w[idx];
}

// ---------------------------------------------------------------------------
// Implicit-GEMM conv via WMMA f32<-f16 16x16x32.
// M = F*OH*OW pixels (16/block), N = COUT (16 per n-tile, NT tiles/wave),
// K = 16*CIN in 32-wide chunks (each chunk = one (kh,kw), 32 channels).
// ---------------------------------------------------------------------------
template <int CIN, int COUT, int IH, int OH, int NT>
__global__ __launch_bounds__(256) void conv_wmma_kernel(
    const _Float16* __restrict__ ain, const _Float16* __restrict__ wt,
    const float* __restrict__ bias, _Float16* __restrict__ aout) {
  constexpr int KTOT = 16 * CIN;
  constexpr int NCHUNK = KTOT / 32;
  constexpr int CPK = CIN / 32;        // chunks per (kh,kw)
  constexpr int OHW = OH * OH;

  __shared__ __align__(16) _Float16 lds_a[16 * 32];

  const int tid  = threadIdx.x;
  const int wave = tid >> 5;
  const int lane = tid & 31;
  const int mtile = blockIdx.x;

  // staging role: each thread stores 2 f16 (row srow, K offset skk)
  const int srow = tid >> 4;
  const int skk  = (tid & 15) << 1;
  const int m_s  = mtile * 16 + srow;
  const int f_s  = m_s / OHW;
  const int rem  = m_s % OHW;
  const int oy_s = rem / OH, ox_s = rem % OH;

  v8f acc[NT];
#pragma unroll
  for (int nt = 0; nt < NT; ++nt)
#pragma unroll
    for (int i = 0; i < 8; ++i) acc[nt][i] = 0.f;

  const int arow  = lane & 15;   // A: M row
  const int khalf = lane >> 4;   // A: K-half select / B: K-run select
  const int ncol  = lane & 15;   // B/D: N column

  for (int chunk = 0; chunk < NCHUNK; ++chunk) {
    const int pos = chunk / CPK;
    const int kh = pos >> 2, kw = pos & 3;
    const int ic = (chunk % CPK) * 32 + skk;
    const int iy = oy_s * 2 + kh - 1;
    const int ix = ox_s * 2 + kw - 1;
    unsigned v = 0u;
    if (iy >= 0 && iy < IH && ix >= 0 && ix < IH)
      v = *(const unsigned*)(ain + ((size_t)(f_s * IH + iy) * IH + ix) * CIN + ic);
    __syncthreads();
    *(unsigned*)(lds_a + srow * 32 + skk) = v;
    __syncthreads();

    // A fragment (16-bit A 16x32 layout): lanes 0-15 rows, K halves interleaved
    v8h alo = *(const v8h*)(lds_a + arow * 32 + khalf * 8);
    v8h ahi = *(const v8h*)(lds_a + arow * 32 + 16 + khalf * 8);
    v16h va;
#pragma unroll
    for (int i = 0; i < 8; ++i) { va[i] = alo[i]; va[8 + i] = ahi[i]; }

    const int kbase = chunk * 32;
#pragma unroll
    for (int nt = 0; nt < NT; ++nt) {
      const int oc = (wave + nt * 8) * 16 + ncol;
      const _Float16* bp = wt + (size_t)oc * KTOT + kbase + khalf * 8;
      v8h blo = *(const v8h*)(bp);
      v8h bhi = *(const v8h*)(bp + 16);
      v16h vb;
#pragma unroll
      for (int i = 0; i < 8; ++i) { vb[i] = blo[i]; vb[8 + i] = bhi[i]; }
      acc[nt] = __builtin_amdgcn_wmma_f32_16x16x32_f16(
          false, va, false, vb, (short)0, acc[nt], false, false);
    }
  }

  // Epilogue: D layout (32-bit C/D 16x16): VGPR r -> M=r (+8 for lanes 16-31)
  const int hi = lane >> 4;
#pragma unroll
  for (int nt = 0; nt < NT; ++nt) {
    const int oc = (wave + nt * 8) * 16 + ncol;
    const float bv = bias[oc];
#pragma unroll
    for (int r = 0; r < 8; ++r) {
      const int m = mtile * 16 + r + 8 * hi;
      const int fo = m / OHW; const int rm = m % OHW;
      const int oy = rm / OH, ox = rm % OH;
      float val = acc[nt][r] + bv;
      val = val > 0.f ? val : 0.f;
      aout[((size_t)(fo * OH + oy) * OH + ox) * COUT + oc] = (_Float16)val;
    }
  }
}

// ---------------------------------------------------------------------------
// Dense: lat[:, :5] only. One block per (frame, n<5); K = 32768 reduction.
// ---------------------------------------------------------------------------
__global__ __launch_bounds__(256) void dense_kernel(
    const _Float16* __restrict__ act3, const float* __restrict__ wd,
    const float* __restrict__ bd, float* __restrict__ a_scr,
    float* __restrict__ dout) {
  __shared__ float red[256];
  const int out = blockIdx.x;        // 0..1919
  const int n = out % 5;
  const int mf = out / 5;
  const _Float16* ap = act3 + (size_t)mf * 32768;
  float s = 0.f;
  for (int k = threadIdx.x; k < 32768; k += 256)
    s += (float)ap[k] * wd[k * 20 + n];
  red[threadIdx.x] = s;
  __syncthreads();
  for (int st = 128; st > 0; st >>= 1) {
    if (threadIdx.x < st) red[threadIdx.x] += red[threadIdx.x + st];
    __syncthreads();
  }
  if (threadIdx.x == 0) {
    float v = red[0] + bd[n];
    a_scr[out] = v;                  // [b*T+t][n]
    dout[OFF_AALL + out] = v;        // a_all output
  }
}

// ---------------------------------------------------------------------------
// LSTM scan, single block, 512 threads (one gate column each), f32.
// Gate order i,f,c,o. Input a_in[t] = (t==0) ? 0 : a_all[:, t-1].
// ---------------------------------------------------------------------------
__global__ __launch_bounds__(512) void lstm_kernel(
    const float* __restrict__ a_scr, const float* __restrict__ lk,
    const float* __restrict__ lrk, const float* __restrict__ lb,
    float* __restrict__ hs) {
  __shared__ float h[B_SZ][128], c[B_SZ][128], g[B_SZ][512], xv[B_SZ][5];
  const int tid = threadIdx.x;
  if (tid < 128) {
    for (int b = 0; b < B_SZ; ++b) { h[b][tid] = 0.f; c[b][tid] = 0.f; }
  }
  __syncthreads();
  for (int t = 0; t < T_SEQ; ++t) {
    if (tid < B_SZ * 5) {
      int b = tid / 5, k = tid % 5;
      xv[b][k] = (t == 0) ? 0.f : a_scr[(b * T_SEQ + t - 1) * 5 + k];
    }
    __syncthreads();
    {
      const int j = tid;
      const float bj = lb[j];
      for (int b = 0; b < B_SZ; ++b) {
        float s = bj;
#pragma unroll
        for (int k = 0; k < 5; ++k) s += xv[b][k] * lk[k * 512 + j];
        for (int k = 0; k < 128; ++k) s += h[b][k] * lrk[k * 512 + j];
        g[b][j] = s;
      }
    }
    __syncthreads();
    if (tid < 128) {
      const int j = tid;
      for (int b = 0; b < B_SZ; ++b) {
        float gi = 1.f / (1.f + expf(-g[b][j]));
        float gf = 1.f / (1.f + expf(-g[b][128 + j]));
        float gg = tanhf(g[b][256 + j]);
        float go = 1.f / (1.f + expf(-g[b][384 + j]));
        float cn = gf * c[b][j] + gi * gg;
        float hn = go * tanhf(cn);
        c[b][j] = cn; h[b][j] = hn;
        hs[((size_t)t * B_SZ + b) * 128 + j] = hn;
      }
    }
    __syncthreads();
  }
}

// ---------------------------------------------------------------------------
// abc = hs @ wabc + babc -> writes A [T,B,10,10] and C [T,B,5,10] into d_out
// ---------------------------------------------------------------------------
__global__ __launch_bounds__(256) void abc_kernel(
    const float* __restrict__ hs, const float* __restrict__ wabc,
    const float* __restrict__ babc, float* __restrict__ dout) {
  int idx = blockIdx.x * blockDim.x + threadIdx.x;
  if (idx >= T_SEQ * B_SZ * 150) return;
  const int n = idx % 150;
  const int tb = idx / 150;          // t*3+b
  const float* h = hs + (size_t)tb * 128;
  float s = babc[n];
  for (int k = 0; k < 128; ++k) s += h[k] * wabc[k * 150 + n];
  if (n < 100) dout[OFF_A + tb * 100 + n] = s;
  else         dout[OFF_C + tb * 50 + (n - 100)] = s;
}

// ---------------------------------------------------------------------------
// Kalman filter scan. One block per batch element, 128 threads.
// Saves z_hat, clamped P_hat, pz, pP per step; writes last_z/last_P.
// ---------------------------------------------------------------------------
__global__ __launch_bounds__(128) void kf_kernel(
    const float* __restrict__ a_scr, float* __restrict__ zh_s,
    float* __restrict__ ph_s, float* __restrict__ pz_s,
    float* __restrict__ pp_s, float* __restrict__ dout) {
  const int b = blockIdx.x, tid = threadIdx.x;
  __shared__ float z[10], P[100], Ash[100], Csh[50], zh[10], AP[100], Ph[100];
  __shared__ float rs[5], CP[50], Ssh[25], aug5[50], Sinv[25], PCt[50], Km[50];
  __shared__ float pzv[10], KC[100], pPv[100];

  if (tid < 10) z[tid] = 0.f;
  if (tid < 100) P[tid] = (tid / 10 == tid % 10) ? 1.f : 0.f;
  __syncthreads();

  for (int t = 0; t < T_SEQ; ++t) {
    const int tb = t * B_SZ + b;
    if (tid < 100) Ash[tid] = dout[OFF_A + tb * 100 + tid];
    if (tid < 50)  Csh[tid] = dout[OFF_C + tb * 50 + tid];
    __syncthreads();
    if (tid < 10) {                                  // z_hat = A z
      float s = 0.f;
      for (int j = 0; j < 10; ++j) s += Ash[tid * 10 + j] * z[j];
      zh[tid] = s;
    }
    if (tid < 100) {                                 // AP = A P
      int i = tid / 10, j = tid % 10; float s = 0.f;
      for (int k = 0; k < 10; ++k) s += Ash[i * 10 + k] * P[k * 10 + j];
      AP[tid] = s;
    }
    __syncthreads();
    if (tid < 100) {                                 // P_hat = AP A^T + Q
      int i = tid / 10, j = tid % 10; float s = (i == j) ? 0.08f : 0.f;
      for (int k = 0; k < 10; ++k) s += AP[i * 10 + k] * Ash[j * 10 + k];
      Ph[tid] = s;
    }
    __syncthreads();
    if (tid < 5) {                                   // resid = a - C z_hat
      float s = a_scr[(b * T_SEQ + t) * 5 + tid];
      for (int j = 0; j < 10; ++j) s -= Csh[tid * 10 + j] * zh[j];
      rs[tid] = s;
    }
    if (tid < 50) {                                  // CP = C P_hat
      int i = tid / 10, j = tid % 10; float s = 0.f;
      for (int k = 0; k < 10; ++k) s += Csh[i * 10 + k] * Ph[k * 10 + j];
      CP[tid] = s;
    }
    __syncthreads();
    if (tid < 25) {                                  // S = CP C^T + R
      int i = tid / 5, j = tid % 5; float s = (i == j) ? 0.03f : 0.f;
      for (int k = 0; k < 10; ++k) s += CP[i * 10 + k] * Csh[j * 10 + k];
      Ssh[tid] = s;
    }
    __syncthreads();
    if (tid < 50) {                                  // Gauss-Jordan 5x5
      int i = tid / 10, j = tid % 10;
      aug5[tid] = (j < 5) ? Ssh[i * 5 + j] : ((j - 5 == i) ? 1.f : 0.f);
    }
    __syncthreads();
    for (int k = 0; k < 5; ++k) {
      float aij = 0.f, akk = 1.f, akj = 0.f, aik = 0.f;
      int i = tid / 10, j = tid % 10;
      if (tid < 50) {
        aij = aug5[tid]; akk = aug5[k * 10 + k];
        akj = aug5[k * 10 + j]; aik = aug5[i * 10 + k];
      }
      __syncthreads();
      if (tid < 50) aug5[tid] = (i == k) ? (akj / akk) : (aij - (aik / akk) * akj);
      __syncthreads();
    }
    if (tid < 25) { int i = tid / 5, j = tid % 5; Sinv[tid] = aug5[i * 10 + 5 + j]; }
    if (tid < 50) {                                  // PCt = P_hat C^T
      int i = tid / 5, j = tid % 5; float s = 0.f;
      for (int k = 0; k < 10; ++k) s += Ph[i * 10 + k] * Csh[j * 10 + k];
      PCt[tid] = s;
    }
    __syncthreads();
    if (tid < 50) {                                  // K = PCt Sinv
      int i = tid / 5, j = tid % 5; float s = 0.f;
      for (int k = 0; k < 5; ++k) s += PCt[i * 5 + k] * Sinv[k * 5 + j];
      Km[tid] = s;
    }
    __syncthreads();
    if (tid < 10) {                                  // pz = z_hat + K resid
      float s = zh[tid];
      for (int j = 0; j < 5; ++j) s += Km[tid * 5 + j] * rs[j];
      pzv[tid] = s;
    }
    if (tid < 100) {                                 // KC = K C
      int i = tid / 10, j = tid % 10; float s = 0.f;
      for (int k = 0; k < 5; ++k) s += Km[i * 5 + k] * Csh[k * 10 + j];
      KC[tid] = s;
    }
    __syncthreads();
    if (tid < 100) {                                 // pP = (I - KC) P_hat
      int i = tid / 10, j = tid % 10; float s = 0.f;
      for (int k = 0; k < 10; ++k)
        s += (((i == k) ? 1.f : 0.f) - KC[i * 10 + k]) * Ph[k * 10 + j];
      pPv[tid] = s;
    }
    __syncthreads();
    if (tid < 10) { zh_s[tb * 10 + tid] = zh[tid]; pz_s[tb * 10 + tid] = pzv[tid]; }
    if (tid < 100) {
      int i = tid / 10, j = tid % 10;
      float cl = (i == j) ? 1e-4f : 0.f;             // max(P_hat, std_min)
      ph_s[tb * 100 + tid] = Ph[tid] > cl ? Ph[tid] : cl;
      pp_s[tb * 100 + tid] = pPv[tid];
    }
    __syncthreads();
    if (tid < 10) z[tid] = pzv[tid];
    if (tid < 100) P[tid] = pPv[tid];
    __syncthreads();
  }
  if (tid < 10)  dout[OFF_LZ + b * 10 + tid] = z[tid];
  if (tid < 100) dout[OFF_LP + b * 100 + tid] = P[tid];
}

// ---------------------------------------------------------------------------
// RTS smoother, backward scan. One block per batch element, 256 threads.
// ---------------------------------------------------------------------------
__global__ __launch_bounds__(256) void rts_kernel(
    const float* __restrict__ zh_s, const float* __restrict__ ph_s,
    const float* __restrict__ pz_s, const float* __restrict__ pp_s,
    float* __restrict__ dout) {
  const int b = blockIdx.x, tid = threadIdx.x;
  __shared__ float zs[10], Ps[100], Al[100], zhl[10], Phl[100], pzl[10], pPl[100];
  __shared__ float aug[200], PhInv[100], t1[100], D[100], Mm[100], DM[100];
  __shared__ float znl[10], Pnl[100];

  if (tid < 10)  zs[tid] = pz_s[((T_SEQ - 1) * B_SZ + b) * 10 + tid];
  if (tid < 100) Ps[tid] = pp_s[((T_SEQ - 1) * B_SZ + b) * 100 + tid];
  __syncthreads();

  for (int t = T_SEQ - 2; t >= 0; --t) {
    const int tb1 = (t + 1) * B_SZ + b, tb0 = t * B_SZ + b;
    if (tid < 100) {
      Al[tid]  = dout[OFF_A + tb1 * 100 + tid];
      Phl[tid] = ph_s[tb1 * 100 + tid];
      pPl[tid] = pp_s[tb0 * 100 + tid];
    }
    if (tid < 10) { zhl[tid] = zh_s[tb1 * 10 + tid]; pzl[tid] = pz_s[tb0 * 10 + tid]; }
    __syncthreads();
    if (tid < 200) {                                 // Gauss-Jordan 10x10 setup
      int i = tid / 20, j = tid % 20;
      aug[tid] = (j < 10) ? Phl[i * 10 + j] : ((j - 10 == i) ? 1.f : 0.f);
    }
    __syncthreads();
    for (int k = 0; k < 10; ++k) {
      float aij = 0.f, akk = 1.f, akj = 0.f, aik = 0.f;
      int i = tid / 20, j = tid % 20;
      if (tid < 200) {
        aij = aug[tid]; akk = aug[k * 20 + k];
        akj = aug[k * 20 + j]; aik = aug[i * 20 + k];
      }
      __syncthreads();
      if (tid < 200) aug[tid] = (i == k) ? (akj / akk) : (aij - (aik / akk) * akj);
      __syncthreads();
    }
    if (tid < 100) { int i = tid / 10, j = tid % 10; PhInv[tid] = aug[i * 20 + 10 + j]; }
    __syncthreads();
    if (tid < 100) {                                 // t1 = pP A^T
      int i = tid / 10, j = tid % 10; float s = 0.f;
      for (int k = 0; k < 10; ++k) s += pPl[i * 10 + k] * Al[j * 10 + k];
      t1[tid] = s;
    }
    __syncthreads();
    if (tid < 100) {                                 // D = t1 PhInv
      int i = tid / 10, j = tid % 10; float s = 0.f;
      for (int k = 0; k < 10; ++k) s += t1[i * 10 + k] * PhInv[k * 10 + j];
      D[tid] = s;
    }
    __syncthreads();
    if (tid < 10) {                                  // zn = pz + D (zs - zh)
      float s = pzl[tid];
      for (int j = 0; j < 10; ++j) s += D[tid * 10 + j] * (zs[j] - zhl[j]);
      znl[tid] = s;
    }
    if (tid < 100) Mm[tid] = Ps[tid] - Phl[tid];
    __syncthreads();
    if (tid < 100) {                                 // DM = D (Ps - Ph)
      int i = tid / 10, j = tid % 10; float s = 0.f;
      for (int k = 0; k < 10; ++k) s += D[i * 10 + k] * Mm[k * 10 + j];
      DM[tid] = s;
    }
    __syncthreads();
    if (tid < 100) {                                 // Pn = pP + DM D^T
      int i = tid / 10, j = tid % 10; float s = pPl[tid];
      for (int k = 0; k < 10; ++k) s += DM[i * 10 + k] * D[j * 10 + k];
      Pnl[tid] = s;
    }
    __syncthreads();
    if (tid < 10) {
      dout[OFF_ZS + (b * (T_SEQ - 1) + t) * 10 + tid] = znl[tid];
      zs[tid] = znl[tid];
    }
    if (tid < 100) {
      dout[OFF_PS + (b * (T_SEQ - 1) + t) * 100 + tid] = Pnl[tid];
      Ps[tid] = Pnl[tid];
    }
    __syncthreads();
  }
}

// ---------------------------------------------------------------------------
// Host launcher
// ---------------------------------------------------------------------------
extern "C" void kernel_launch(void* const* d_in, const int* in_sizes, int n_in,
                              void* d_out, int out_size, void* d_ws, size_t ws_size,
                              hipStream_t stream) {
  (void)in_sizes; (void)n_in; (void)out_size; (void)ws_size;
  const float* img  = (const float*)d_in[0];
  const float* w1   = (const float*)d_in[1];
  const float* b1   = (const float*)d_in[2];
  const float* w2   = (const float*)d_in[3];
  const float* b2   = (const float*)d_in[4];
  const float* w3   = (const float*)d_in[5];
  const float* b3   = (const float*)d_in[6];
  const float* wd   = (const float*)d_in[7];
  const float* bd   = (const float*)d_in[8];
  const float* lk   = (const float*)d_in[9];
  const float* lrk  = (const float*)d_in[10];
  const float* lb   = (const float*)d_in[11];
  const float* wabc = (const float*)d_in[12];
  const float* babc = (const float*)d_in[13];
  float* out = (float*)d_out;

  char* ws = (char*)d_ws;
  size_t off = 0;
  auto alloc_b = [&](size_t bytes) {
    size_t r = off; off += (bytes + 255) & ~(size_t)255; return r;
  };
  _Float16* act1 = (_Float16*)(ws + alloc_b(25165824ull * 2)); // 384*32*32*64
  _Float16* act2 = (_Float16*)(ws + alloc_b(12582912ull * 2)); // 384*16*16*128
  _Float16* act3 = (_Float16*)(ws + alloc_b(12582912ull * 2)); // 384*8*8*512
  _Float16* wt2  = (_Float16*)(ws + alloc_b(131072ull * 2));   // 128 x 1024
  _Float16* wt3  = (_Float16*)(ws + alloc_b(1048576ull * 2));  // 512 x 2048
  float* a_scr = (float*)(ws + alloc_b(1920ull * 4));
  float* hs    = (float*)(ws + alloc_b(49152ull * 4));
  float* zh_s  = (float*)(ws + alloc_b(3840ull * 4));
  float* ph_s  = (float*)(ws + alloc_b(38400ull * 4));
  float* pz_s  = (float*)(ws + alloc_b(3840ull * 4));
  float* pp_s  = (float*)(ws + alloc_b(38400ull * 4));

  conv1_kernel<<<98304, 256, 0, stream>>>(img, w1, b1, act1);
  wprep_kernel<64, 128><<<512, 256, 0, stream>>>(w2, wt2);
  wprep_kernel<128, 512><<<4096, 256, 0, stream>>>(w3, wt3);
  conv_wmma_kernel<64, 128, 32, 16, 1><<<6144, 256, 0, stream>>>(act1, wt2, b2, act2);
  conv_wmma_kernel<128, 512, 16, 8, 4><<<1536, 256, 0, stream>>>(act2, wt3, b3, act3);
  dense_kernel<<<1920, 256, 0, stream>>>(act3, wd, bd, a_scr, out);
  lstm_kernel<<<1, 512, 0, stream>>>(a_scr, lk, lrk, lb, hs);
  abc_kernel<<<225, 256, 0, stream>>>(hs, wabc, babc, out);
  kf_kernel<<<3, 128, 0, stream>>>(a_scr, zh_s, ph_s, pz_s, pp_s, out);
  rts_kernel<<<3, 256, 0, stream>>>(zh_s, ph_s, pz_s, pp_s, out);
}
